// SparseResidualBlock_84155589198508
// MI455X (gfx1250) — matrix operand order repaired and checked by reading, not compiled, not run on hardware
//
#include <hip/hip_runtime.h>

// ---------------------------------------------------------------------------
// SparseResidualBlock on gfx1250 (MI455X)
// gather-GEMM via v_wmma_f32_16x16x32_bf16, weights staged in 216KB LDS,
// double-buffered B fragments (ds) + 2-stage pipelined A gather (vmem)
// ---------------------------------------------------------------------------

typedef __attribute__((ext_vector_type(16))) __bf16 v16bf;
typedef __attribute__((ext_vector_type(8)))  float  v8f;

#define NVOX  120000
#define CCH   64
#define KOFF  27
#define TILES32 (NVOX / 32)                    // 3750, exact
#define WPACK_ELEMS (KOFF * 2 * 4 * 2 * 32 * 8)   // 110592 bf16 per layer
#define WPACK_BYTES (WPACK_ELEMS * 2)             // 221184 B
#define WPACK_U4    (WPACK_BYTES / 16)            // 13824 uint4

static __device__ __forceinline__ unsigned short f32_bf16(float f) {
  unsigned int x = __float_as_uint(f);
  unsigned int r = x + 0x7fffu + ((x >> 16) & 1u);   // round to nearest even
  return (unsigned short)(r >> 16);
}

union Frag {                 // 32 bytes: one WMMA A/B operand per lane
  uint4 u[2];
  v16bf v;
};

struct APair { Frag f[2][2]; };   // [mtile][kt]

// ---- prep: f32 -> bf16 elementwise -----------------------------------------
__global__ __launch_bounds__(256) void cvt_bf16_kernel(
    const float* __restrict__ src, unsigned short* __restrict__ dst, int n) {
  int i = blockIdx.x * 256 + threadIdx.x;
  if (i < n) dst[i] = f32_bf16(src[i]);
}

// ---- prep: pack W[27,64,64] f32 into LDS-ready B-fragment layout -----------
// Wp layout: [k][kt(2)][nt(4)][half(2)][lane(32)][e2(8)]  (16B per lane chunk
// -> ds_load_b128 lane stride 16B: conflict-free across 64 banks)
//   element = W[k][ cin = kt*32 + (lane>=16?8:0) + half*16 + e2 ]
//              [ cout = nt*16 + (lane&15) ]
__global__ __launch_bounds__(256) void pack_w_kernel(
    const float* __restrict__ W, unsigned short* __restrict__ Wp) {
  int i = blockIdx.x * 256 + threadIdx.x;
  if (i >= WPACK_ELEMS) return;
  int e2   = i & 7;
  int lane = (i >> 3) & 31;
  int h    = (i >> 8) & 1;
  int nt   = (i >> 9) & 3;
  int kt   = (i >> 11) & 1;
  int k    = i >> 12;
  int cin  = kt * 32 + ((lane >> 4) << 3) + h * 16 + e2;
  int cout = nt * 16 + (lane & 15);
  Wp[i] = f32_bf16(W[(k * CCH + cin) * CCH + cout]);
}

// ---- A-fragment gather (ISA 16-bit 16x32 A layout), M=32 = 2 m-tiles -------
static __device__ __forceinline__ void load_a(
    APair& A, const unsigned short* __restrict__ Ain,
    int idx0, int idx1, int hi) {
#pragma unroll
  for (int mt = 0; mt < 2; ++mt) {
    const int id = mt ? idx1 : idx0;
#pragma unroll
    for (int kt = 0; kt < 2; ++kt) {
      if (id >= 0) {
        const unsigned short* arow = Ain + (long)id * CCH + kt * 32 + hi * 8;
        A.f[mt][kt].u[0] = *reinterpret_cast<const uint4*>(arow);
        A.f[mt][kt].u[1] = *reinterpret_cast<const uint4*>(arow + 16);
      } else {
        A.f[mt][kt].u[0] = make_uint4(0u, 0u, 0u, 0u);
        A.f[mt][kt].u[1] = make_uint4(0u, 0u, 0u, 0u);
      }
    }
  }
}

// ---- 16 WMMAs for one k-offset: B double-buffered from LDS -----------------
// Fragment f = kt*4 + nt; loads for f+1 issue BEFORE the WMMAs of f so the
// ds latency overlaps the matrix pipe (partial dscnt waits).
static __device__ __forceinline__ void compute_k(
    int k, const APair& A, const uint4* __restrict__ s4, int lane,
    v8f acc0[4], v8f acc1[4]) {
  Frag B[2];
  {
    const int fid0 = k * 8;                       // f = 0
    B[0].u[0] = s4[(fid0 * 2 + 0) * 32 + lane];
    B[0].u[1] = s4[(fid0 * 2 + 1) * 32 + lane];
  }
#pragma unroll
  for (int f = 0; f < 8; ++f) {
    if (f + 1 < 8) {                              // prefetch next B fragment
      const int fid = k * 8 + f + 1;
      B[(f + 1) & 1].u[0] = s4[(fid * 2 + 0) * 32 + lane];
      B[(f + 1) & 1].u[1] = s4[(fid * 2 + 1) * 32 + lane];
    }
    const int kt = f >> 2;
    const int nt = f & 3;
    acc0[nt] = __builtin_amdgcn_wmma_f32_16x16x32_bf16(
        false, A.f[0][kt].v, false, B[f & 1].v, (short)0, acc0[nt], false, false);
    acc1[nt] = __builtin_amdgcn_wmma_f32_16x16x32_bf16(
        false, A.f[1][kt].v, false, B[f & 1].v, (short)0, acc1[nt], false, false);
  }
}

// ---- main gather-GEMM: one wave32 per 32-voxel x 64-channel tile -----------
template <int LAYER>
__global__ __launch_bounds__(512) void subm_gemm_kernel(
    const unsigned short* __restrict__ Ain,   // bf16 activations [NVOX][64]
    const int*            __restrict__ nbr,   // [NVOX][27]
    const unsigned short* __restrict__ Wp,    // packed bf16 weights (global)
    const float* __restrict__ gn, const float* __restrict__ bt,
    const float* __restrict__ mu, const float* __restrict__ vr,
    const float* __restrict__ resid,          // f32 features (layer 2 only)
    unsigned short* __restrict__ out_bf,      // layer-1 output (bf16)
    float*          __restrict__ out_f)       // layer-2 output (f32)
{
  extern __shared__ char smem_raw[];          // 221184 B: packed weights
  uint4* s4 = reinterpret_cast<uint4*>(smem_raw);

  // --- stage all 27x(2x4) B fragments into LDS once per workgroup ---
  {
    const uint4* g4 = reinterpret_cast<const uint4*>(Wp);
#pragma unroll 4
    for (int i = threadIdx.x; i < WPACK_U4; i += 512) s4[i] = g4[i];
  }
  __syncthreads();

  const int wave = threadIdx.x >> 5;
  const int lane = threadIdx.x & 31;
  const int tile = blockIdx.x * 16 + wave;
  if (tile >= TILES32) return;                // wave-uniform

  const int v0 = tile * 32;
  const int m  = lane & 15;
  const int hi = lane >> 4;

  v8f acc0[4] = {};
  v8f acc1[4] = {};

  const int* nrow0 = nbr + (long)(v0 + m) * KOFF;
  const int* nrow1 = nbr + (long)(v0 + 16 + m) * KOFF;

  // --- 2-stage software pipeline over the 27 kernel offsets ---
  APair Aa, Ab;
  {
    int i0 = nrow0[0], i1 = nrow1[0];
    load_a(Aa, Ain, i0, i1, hi);
  }
  int p0 = nrow0[1], p1 = nrow1[1];           // idx for k=1

#pragma unroll 1
  for (int k = 0; k < KOFF; k += 2) {
    if (k + 1 < KOFF) load_a(Ab, Ain, p0, p1, hi);   // gather A(k+1)
    int q0 = 0, q1 = 0;
    if (k + 2 < KOFF) { q0 = nrow0[k + 2]; q1 = nrow1[k + 2]; }
    compute_k(k, Aa, s4, lane, acc0, acc1);
    if (k + 1 < KOFF) {
      if (k + 2 < KOFF) load_a(Aa, Ain, q0, q1, hi); // gather A(k+2)
      if (k + 3 < KOFF) { p0 = nrow0[k + 3]; p1 = nrow1[k + 3]; }
      compute_k(k + 1, Ab, s4, lane, acc0, acc1);
    }
  }

  // --- fused BN (+ residual) + ReLU epilogue ---
#pragma unroll
  for (int nt = 0; nt < 4; ++nt) {
    const int   c    = nt * 16 + m;
    const float sc   = gn[c] * rsqrtf(vr[c] + 1e-5f);
    const float mean = mu[c];
    const float beta = bt[c];
#pragma unroll
    for (int j = 0; j < 8; ++j) {             // C/D VGPR j -> M = j + hi*8
      const long r0 = v0 + j + hi * 8;
      const long r1 = r0 + 16;
      float val0 = (acc0[nt][j] - mean) * sc + beta;
      float val1 = (acc1[nt][j] - mean) * sc + beta;
      if (LAYER == 1) {
        val0 = val0 > 0.f ? val0 : 0.f;
        val1 = val1 > 0.f ? val1 : 0.f;
        out_bf[r0 * CCH + c] = f32_bf16(val0);
        out_bf[r1 * CCH + c] = f32_bf16(val1);
      } else {
        val0 += resid[r0 * CCH + c];
        val1 += resid[r1 * CCH + c];
        val0 = val0 > 0.f ? val0 : 0.f;
        val1 = val1 > 0.f ? val1 : 0.f;
        out_f[r0 * CCH + c] = val0;
        out_f[r1 * CCH + c] = val1;
      }
    }
  }
}

// ---------------------------------------------------------------------------
extern "C" void kernel_launch(void* const* d_in, const int* in_sizes, int n_in,
                              void* d_out, int out_size, void* d_ws, size_t ws_size,
                              hipStream_t stream) {
  (void)in_sizes; (void)n_in; (void)out_size; (void)ws_size;

  const float* features = (const float*)d_in[0];
  const float* W1 = (const float*)d_in[1];
  const float* g1 = (const float*)d_in[2];
  const float* b1 = (const float*)d_in[3];
  const float* m1 = (const float*)d_in[4];
  const float* v1 = (const float*)d_in[5];
  const float* W2 = (const float*)d_in[6];
  const float* g2 = (const float*)d_in[7];
  const float* b2 = (const float*)d_in[8];
  const float* m2 = (const float*)d_in[9];
  const float* v2 = (const float*)d_in[10];
  const int*  nbr = (const int*)d_in[11];
  float*      out = (float*)d_out;

  // workspace layout (256-byte aligned slabs)
  char* ws = (char*)d_ws;
  const size_t featBytes = (size_t)NVOX * CCH * sizeof(unsigned short);
  const size_t featAl = (featBytes + 255) & ~(size_t)255;
  const size_t wAl = ((size_t)WPACK_BYTES + 255) & ~(size_t)255;
  unsigned short* featb = (unsigned short*)ws;
  unsigned short* midb  = (unsigned short*)(ws + featAl);
  unsigned short* W1p   = (unsigned short*)(ws + 2 * featAl);
  unsigned short* W2p   = (unsigned short*)(ws + 2 * featAl + wAl);

  // 1) convert activations to bf16
  {
    int n = NVOX * CCH;
    cvt_bf16_kernel<<<(n + 255) / 256, 256, 0, stream>>>(features, featb, n);
  }
  // 2) pack both weight tensors into the LDS-ready fragment layout
  pack_w_kernel<<<(WPACK_ELEMS + 255) / 256, 256, 0, stream>>>(W1, W1p);
  pack_w_kernel<<<(WPACK_ELEMS + 255) / 256, 256, 0, stream>>>(W2, W2p);

  // 3) conv1 + BN + ReLU -> bf16 intermediate
  const int blocks = (TILES32 + 15) / 16;   // 16 waves (M32-tiles) per block
  subm_gemm_kernel<1><<<blocks, 512, WPACK_BYTES, stream>>>(
      featb, nbr, W1p, g1, b1, m1, v1, nullptr, midb, nullptr);

  // 4) conv2 + BN + residual + ReLU -> f32 output
  subm_gemm_kernel<2><<<blocks, 512, WPACK_BYTES, stream>>>(
      midb, nbr, W2p, g2, b2, m2, v2, features, nullptr, out);
}
